// seqKAN_59794534695203
// MI455X (gfx1250) — matrix operand order repaired and checked by reading, not compiled, not run on hardware
//
#include <hip/hip_runtime.h>

// ---------------------------------------------------------------------------
// seqKAN on MI455X (gfx1250): KAN layer == GEMM with K = 640 (silu) + 5120
// (bspline basis) = 5760.  bf16 WMMA (v_wmma_f32_16x16x32_bf16), data staged
// in the exact CDNA5 WMMA fragment layouts so the GEMM loop is pure
// load_b128 + wmma, software-pipelined one K-tile ahead.
// ---------------------------------------------------------------------------

#define IN_   128
#define HID_  512
#define OUT_  128
#define BATCH_ 256
#define T_    128
#define F_    (IN_ + HID_)          // 640
#define NB_   8                     // G + K = 5 + 3
#define KDIM  (F_ + F_ * NB_)       // 5760
#define NKT   (KDIM / 32)           // 180 K-tiles of 32

typedef __bf16 v16bf __attribute__((ext_vector_type(16)));
typedef float  v8f   __attribute__((ext_vector_type(8)));

__device__ __forceinline__ unsigned short f2bf(float f) {
    unsigned int u = __float_as_uint(f);
    u = (u + 0x7FFFu + ((u >> 16) & 1u)) >> 16;     // round-to-nearest-even
    return (unsigned short)u;
}

// A-fragment address helper: 16-bit A 16x32 layout (ISA 7.12.2).
// lane = m + (kk&8 ? 16 : 0); element e = (kk&7) + (kk&16 ? 8 : 0).
// Buffer: v16bf slot index = (mTile*NKT + kTile)*32 + lane, 16 bf16 per slot.
__device__ __forceinline__ int afrag_base(int mTile, int kTile, int lane) {
    return (((mTile * NKT + kTile) << 5) + lane) << 4;   // in bf16 elements
}

// ---------------------------------------------------------------------------
// Build combined weight matrix in 16-bit B-matrix (32x16, K x N) fragment
// layout: lane = n%16 + (k&8 ? 16 : 0), e = (k&7) + (k&16 ? 8 : 0), one v16bf slot per (nTile,kTile,lane).
//   k <  640            : W[k,o] = w_base[k,o]
//   k = 640 + f*8 + g   : W[k,o] = coef[f,o,g] * w_sp[f,o]
// ---------------------------------------------------------------------------
__global__ void __launch_bounds__(256) prep_weights(
    const float* __restrict__ w_base, const float* __restrict__ w_sp,
    const float* __restrict__ coef, unsigned short* __restrict__ Wfrag,
    int N, int total) {
    int idx = blockIdx.x * 256 + threadIdx.x;
    if (idx >= total) return;
    int k = idx / N;
    int o = idx - k * N;
    float v;
    if (k < F_) {
        v = w_base[k * N + o];
    } else {
        int r = k - F_;
        int f = r >> 3, g = r & 7;
        v = coef[((size_t)(f * N + o) << 3) + g] * w_sp[f * N + o];
    }
    int kT = k >> 5, kk = k & 31, nT = o >> 4;
    int lane = (o & 15) + ((kk & 8) ? 16 : 0), e = (kk & 7) + ((kk & 16) ? 8 : 0);
    Wfrag[((((nT * NKT + kT) << 5) + lane) << 4) + e] = f2bf(v);
}

// ---------------------------------------------------------------------------
// Featurize one timestep: for each (b, f) compute silu + 8 cubic B-spline
// basis values (uniform knots t_i = -2.2 + 0.4*i), write bf16 directly in
// A-fragment layout.  The 8 basis values of one f land in ONE lane at a
// contiguous element range -> single 16-byte packed store.
// ---------------------------------------------------------------------------
__global__ void __launch_bounds__(256) kan_featurize(
    const float* __restrict__ x, int t, const float* __restrict__ h,
    unsigned short* __restrict__ Afrag) {
    int idx = blockIdx.x * 256 + threadIdx.x;        // BATCH_*F_ = 163840
    int b = idx / F_;
    int f = idx - b * F_;

    float v = (f < IN_) ? x[((size_t)b * T_ + t) * IN_ + f]
                        : h[(size_t)b * HID_ + (f - IN_)];

    // SiLU
    float s = v * (1.0f / (1.0f + __expf(-v)));

    // Cox-de Boor, degree 0..3, uniform knots, h = 0.4, t0 = -2.2
    float bb[11];
#pragma unroll
    for (int j = 0; j < 11; ++j) {
        float tj = -2.2f + 0.4f * (float)j;
        bb[j] = (v >= tj && v < tj + 0.4f) ? 1.0f : 0.0f;
    }
#pragma unroll
    for (int p = 1; p <= 3; ++p) {
        float inv = 1.0f / (0.4f * (float)p);
#pragma unroll
        for (int j = 0; j < 10; ++j) {
            if (j < 11 - p) {
                float tj    = -2.2f + 0.4f * (float)j;
                float left  = (v - tj) * inv * bb[j];
                float right = (tj + 0.4f * (float)(p + 1) - v) * inv * bb[j + 1];
                bb[j] = left + right;
            }
        }
    }

    int mTile = b >> 4, m = b & 15;

    // silu value at column k = f
    {
        int k = f, kT = k >> 5, kk = k & 31;
        int lane = m + ((kk & 8) ? 16 : 0);
        int e    = (kk & 7) + ((kk & 16) ? 8 : 0);
        Afrag[afrag_base(mTile, kT, lane) + e] = f2bf(s);
    }
    // basis values at columns k0 = 640 + f*8 .. +7 (one lane, contiguous e)
    {
        int k0 = F_ + (f << 3);
        int kT = k0 >> 5, kk = k0 & 31;               // kk in {0,8,16,24}
        int lane = m + ((kk & 8) ? 16 : 0);
        int e0   = (kk & 16) ? 8 : 0;
        unsigned int p0 = (unsigned)f2bf(bb[0]) | ((unsigned)f2bf(bb[1]) << 16);
        unsigned int p1 = (unsigned)f2bf(bb[2]) | ((unsigned)f2bf(bb[3]) << 16);
        unsigned int p2 = (unsigned)f2bf(bb[4]) | ((unsigned)f2bf(bb[5]) << 16);
        unsigned int p3 = (unsigned)f2bf(bb[6]) | ((unsigned)f2bf(bb[7]) << 16);
        uint4 pk; pk.x = p0; pk.y = p1; pk.z = p2; pk.w = p3;
        *reinterpret_cast<uint4*>(&Afrag[afrag_base(mTile, kT, lane) + e0]) = pk;
    }
}

// ---------------------------------------------------------------------------
// GEMM: out[256, N] = A[256, 5760] x W[5760, N], bf16 in / f32 out.
// Block = 256 threads = 8 waves.  Block tile 64(M) x 64(N); each wave owns
// one 16x16 M-tile x two N-tiles (2 accumulators).
// Software-pipelined: fragments for K-tile k+1 are loaded before the WMMAs
// of K-tile k issue, so the waitcnt inserter can overlap the 6 outstanding
// b128 loads with the matrix math instead of draining loadcnt to 0.
// Unconditional global_prefetch 16 K-tiles (16 KB) ahead keeps the streams
// resident in the WGP cache (speculative; over-run lands in adjacent ws).
// ---------------------------------------------------------------------------
__global__ void __launch_bounds__(256) kan_gemm(
    const unsigned short* __restrict__ Afrag,
    const unsigned short* __restrict__ Wfrag,
    float* __restrict__ out, int N) {
    int lane = threadIdx.x & 31;
    int wave = threadIdx.x >> 5;
    int mTile  = blockIdx.y * 4 + (wave >> 1);        // 0..15
    int nTile0 = blockIdx.x * 4 + (wave & 1) * 2;     // pairs of N-tiles

    const v16bf* Ap  = reinterpret_cast<const v16bf*>(Afrag)
                       + ((size_t)mTile * NKT << 5) + lane;
    const v16bf* Bp0 = reinterpret_cast<const v16bf*>(Wfrag)
                       + ((size_t)nTile0 * NKT << 5) + lane;
    const v16bf* Bp1 = Bp0 + ((size_t)NKT << 5);

    v8f acc0 = {}; v8f acc1 = {};

    // prologue: stage K-tile 0
    v16bf a  = Ap[0];
    v16bf b0 = Bp0[0];
    v16bf b1 = Bp1[0];

    for (int k = 1; k < NKT; ++k) {
        size_t off = (size_t)k << 5;
        // issue next K-tile's loads first (overlaps with this tile's WMMAs)
        v16bf an  = Ap [off];
        v16bf b0n = Bp0[off];
        v16bf b1n = Bp1[off];
        // stream prefetch 16 K-tiles ahead (speculative, no guard branches)
        __builtin_prefetch((const void*)(Ap  + off + (16u << 5)), 0, 3);
        __builtin_prefetch((const void*)(Bp0 + off + (16u << 5)), 0, 3);
        __builtin_prefetch((const void*)(Bp1 + off + (16u << 5)), 0, 3);

        acc0 = __builtin_amdgcn_wmma_f32_16x16x32_bf16(
                   false, a, false, b0, (short)0, acc0, false, false);
        acc1 = __builtin_amdgcn_wmma_f32_16x16x32_bf16(
                   false, a, false, b1, (short)0, acc1, false, false);
        a = an; b0 = b0n; b1 = b1n;
    }
    // epilogue: last K-tile
    acc0 = __builtin_amdgcn_wmma_f32_16x16x32_bf16(
               false, a, false, b0, (short)0, acc0, false, false);
    acc1 = __builtin_amdgcn_wmma_f32_16x16x32_bf16(
               false, a, false, b1, (short)0, acc1, false, false);

    // f32 C/D layout: element r -> row r + (lane>=16 ? 8 : 0), col = lane&15
    int nCol = lane & 15;
    int mRow = mTile * 16 + ((lane >> 4) << 3);
    float* op = out + (size_t)mRow * N + nTile0 * 16 + nCol;
#pragma unroll
    for (int r = 0; r < 8; ++r) {
        op[(size_t)r * N]      = acc0[r];
        op[(size_t)r * N + 16] = acc1[r];
    }
}

// ---------------------------------------------------------------------------
extern "C" void kernel_launch(void* const* d_in, const int* in_sizes, int n_in,
                              void* d_out, int out_size, void* d_ws, size_t ws_size,
                              hipStream_t stream) {
    const float* x        = (const float*)d_in[0];
    const float* w_base_h = (const float*)d_in[1];
    const float* w_sp_h   = (const float*)d_in[2];
    const float* coef_h   = (const float*)d_in[3];
    const float* w_base_o = (const float*)d_in[4];
    const float* w_sp_o   = (const float*)d_in[5];
    const float* coef_o   = (const float*)d_in[6];
    float* out = (float*)d_out;

    // workspace layout (bytes)
    const size_t szAfrag = (size_t)BATCH_ * KDIM * 2;        // 2,949,120
    const size_t szWh    = (size_t)KDIM * HID_ * 2;          // 5,898,240
    const size_t szWo    = (size_t)KDIM * OUT_ * 2;          // 1,474,560
    const size_t szH     = (size_t)BATCH_ * HID_ * 4;        //   524,288
    char* ws = (char*)d_ws;
    unsigned short* Afrag = (unsigned short*)ws;
    unsigned short* Wh    = (unsigned short*)(ws + szAfrag);
    unsigned short* Wo    = (unsigned short*)(ws + szAfrag + szWh);
    float* hA             = (float*)(ws + szAfrag + szWh + szWo);
    float* hB             = (float*)(ws + szAfrag + szWh + szWo + szH);

    // 1) fragment-swizzled combined weights (cheap, once per launch)
    {
        int tot = KDIM * HID_;
        prep_weights<<<(tot + 255) / 256, 256, 0, stream>>>(
            w_base_h, w_sp_h, coef_h, Wh, HID_, tot);
        tot = KDIM * OUT_;
        prep_weights<<<(tot + 255) / 256, 256, 0, stream>>>(
            w_base_o, w_sp_o, coef_o, Wo, OUT_, tot);
    }
    // 2) h0 = 0
    hipMemsetAsync(hA, 0, szH, stream);

    // 3) recurrence: h_t = KAN_h([x_t, h_{t-1}])
    float* hin = hA; float* hout = hB;
    for (int t = 0; t < T_; ++t) {
        kan_featurize<<<(BATCH_ * F_) / 256, 256, 0, stream>>>(x, t, hin, Afrag);
        kan_gemm<<<dim3(HID_ / 64, BATCH_ / 64), 256, 0, stream>>>(
            Afrag, Wh, hout, HID_);
        float* tmp = hin; hin = hout; hout = tmp;
    }

    // 4) final output layer only: out = KAN_o([x_{T-2}, h_{T-1}])
    kan_featurize<<<(BATCH_ * F_) / 256, 256, 0, stream>>>(x, T_ - 2, hin, Afrag);
    kan_gemm<<<dim3(OUT_ / 64, BATCH_ / 64), 256, 0, stream>>>(
        Afrag, Wo, out, OUT_);
}
